// Attention_36919538876665
// MI455X (gfx1250) — matrix operand order repaired and checked by reading, not compiled, not run on hardware
//
#include <hip/hip_runtime.h>
#include <hip/hip_bf16.h>

// ---------------------------------------------------------------------------
// Performer causal linear attention for MI455X (gfx1250, wave32, WMMA bf16,
// TDM-staged prefix state). All matmuls: v_wmma_f32_16x16x32_bf16.
// ---------------------------------------------------------------------------

typedef __attribute__((ext_vector_type(16))) __bf16    v16bf;
typedef __attribute__((ext_vector_type(8)))  float     v8f;
typedef __attribute__((ext_vector_type(4)))  unsigned  v4u;
typedef __attribute__((ext_vector_type(8)))  int       v8i;
typedef __attribute__((ext_vector_type(4)))  int       v4i;

#if defined(__has_builtin)
#if __has_builtin(__builtin_amdgcn_tensor_load_to_lds) && __has_builtin(__builtin_amdgcn_s_wait_tensorcnt)
#define HAVE_TDM 1
#endif
#endif
#ifndef HAVE_TDM
#define HAVE_TDM 0
#endif

static constexpr int Bc  = 2;
static constexpr int Hc  = 16;
static constexpr int Nc  = 4096;
static constexpr int Dc  = 64;
static constexpr int Mc  = 256;
static constexpr int CHc = 128;
static constexpr int NCH = Nc / CHc;      // 32 chunks per sequence
static constexpr int BH  = Bc * Hc;       // 32
static constexpr int NBLK = BH * NCH;     // 1024 chunk-blocks

static constexpr float DN    = 0.35355339059327379f;  // 64^-0.25
static constexpr float RATIO = 0.0625f;               // 256^-0.5
static constexpr float KEPS  = 1e-4f;
static constexpr float EPSC  = 1e-6f;

// LDS row strides (bf16 elements), padded +8 to dodge bank conflicts
static constexpr int LD64  = 72;
static constexpr int LD128 = 136;
static constexpr int LD256 = 264;

// workspace: S (bf16, stored transposed [d][m]), Z (f32), global max key
static constexpr size_t S_BYTES = (size_t)NBLK * Mc * Dc * sizeof(__bf16); // 32 MB
static constexpr size_t Z_BYTES = (size_t)NBLK * Mc * sizeof(float);       //  1 MB

__device__ inline v8f vzero8() {
  v8f a;
#pragma unroll
  for (int i = 0; i < 8; ++i) a[i] = 0.0f;
  return a;
}

// order-preserving float <-> uint key (for atomic max)
__device__ inline unsigned fkey(float f) {
  unsigned u = __float_as_uint(f);
  return (u & 0x80000000u) ? ~u : (u | 0x80000000u);
}
__device__ inline float fdec(unsigned kk) {
  unsigned u = (kk & 0x80000000u) ? (kk & 0x7FFFFFFFu) : ~kk;
  return __uint_as_float(u);
}

// A fragment: rows row0..row0+15 of row-major A, K-slice k0..k0+31.
// Two 8xbf16 runs per lane (k 8h.., 16+8h..) -> 2x ds_load_b128. (ISA 7.12.2 layout)
__device__ inline v16bf load_afrag(const __bf16* A, int lda, int row0, int k0) {
  const int lane = threadIdx.x & 31, hi = lane >> 4;
  const int mrow = row0 + (lane & 15);
  v16bf a;
#pragma unroll
  for (int e = 0; e < 16; ++e) {
    int p = e >> 1, bb = e & 1;
    int kA = (p < 4) ? (2 * p + bb + 8 * hi) : (16 + 2 * (p - 4) + bb + 8 * hi);
    a[e] = A[(size_t)mrow * lda + k0 + kA];
  }
  return a;
}

// B fragment from transposed storage Bt[n][k]: table k order (matches A).
__device__ inline v16bf load_bfragT(const __bf16* Bt, int ldb, int col0, int k0) {
  const int lane = threadIdx.x & 31, hi = lane >> 4;
  const int ncol = col0 + (lane & 15);
  v16bf b;
#pragma unroll
  for (int e = 0; e < 16; ++e)
    b[e] = Bt[(size_t)ncol * ldb + k0 + 8 * hi + e + ((e >> 3) << 3)];
  return b;
}

__device__ inline v8f wmma_bf16(v16bf a, v16bf b, v8f c) {
  return __builtin_amdgcn_wmma_f32_16x16x32_bf16(false, a, false, b, (short)0, c,
                                                 false, false);
}

__device__ inline void stage_bf16(__bf16* dst, int ldst, const float* src,
                                  int rows, int cols, float scale) {
  for (int i = threadIdx.x; i < rows * cols; i += blockDim.x) {
    int r = i / cols, c = i % cols;
    dst[(size_t)r * ldst + c] = (__bf16)(src[i] * scale);
  }
}
// stage [rows][cols] f32 -> transposed bf16 [cols][ldst]
__device__ inline void stage_bf16_T(__bf16* dst, int ldst, const float* src,
                                    int rows, int cols) {
  for (int i = threadIdx.x; i < rows * cols; i += blockDim.x) {
    int r = i / cols, c = i % cols;
    dst[(size_t)c * ldst + r] = (__bf16)src[i];
  }
}

// ---------------------------------------------------------------------------
__global__ void perf_init_kernel(unsigned* gkey) {
  if (threadIdx.x == 0 && blockIdx.x == 0) *gkey = 0u;
}

// Pass 1: global max of k's data_dash (k stabilizer is a global scalar max)
__global__ __launch_bounds__(256) void perf_kmax_kernel(const float* __restrict__ kin,
                                                        const float* __restrict__ proj,
                                                        unsigned* gkey) {
  extern __shared__ char smem[];
  __bf16* P  = (__bf16*)smem;               // [256][72] (B^T storage)
  __bf16* Xk = (__bf16*)(smem + 36864);     // [128][72]
  const int bh = blockIdx.x / NCH, c = blockIdx.x % NCH;
  const float* ks = kin + (size_t)(bh * Nc + c * CHc) * Dc;
  stage_bf16(P, LD64, proj, Mc, Dc, 1.0f);
  stage_bf16(Xk, LD64, ks, CHc, Dc, DN);
  __syncthreads();
  const int w = __builtin_amdgcn_readfirstlane((int)(threadIdx.x >> 5));
  // wave w owns row-tile w; hoist A fragments, sweep 16 m-tiles
  v16bf a0 = load_afrag(Xk, LD64, w * 16, 0);
  v16bf a1 = load_afrag(Xk, LD64, w * 16, 32);
  float lm = -3.0e38f;
  for (int tj = 0; tj < 16; ++tj) {
    v8f acc = wmma_bf16(a0, load_bfragT(P, LD64, tj * 16, 0), vzero8());
    acc = wmma_bf16(a1, load_bfragT(P, LD64, tj * 16, 32), acc);
#pragma unroll
    for (int r = 0; r < 8; ++r) lm = fmaxf(lm, acc[r]);
  }
  atomicMax(gkey, fkey(lm));
}

// Pass 2: per-chunk S^T = (kp^T v)^T stored [d][m] bf16, Z = colsum(kp) f32
__global__ __launch_bounds__(256) void perf_sz_kernel(const float* __restrict__ kin,
                                                      const float* __restrict__ vin,
                                                      const float* __restrict__ proj,
                                                      const unsigned* gkey,
                                                      __bf16* __restrict__ gS,
                                                      float* __restrict__ gZ) {
  extern __shared__ char smem[];
  __bf16* KPT = (__bf16*)smem;                  // [256][136]  kp^T : [m][j]
  __bf16* VT  = (__bf16*)(smem + 69632);        // [64][136]   v^T  : [d][j]
  __bf16* P   = (__bf16*)(smem + 87040);        // [256][72]
  __bf16* Xk  = (__bf16*)(smem + 123904);       // [128][72]
  float* diagk = (float*)(smem + 142336);       // [128]
  const int bh = blockIdx.x / NCH, c = blockIdx.x % NCH;
  const int tid = threadIdx.x, lane = tid & 31, hi = lane >> 4;
  const int w = __builtin_amdgcn_readfirstlane((int)(tid >> 5));
  const size_t base = (size_t)(bh * Nc + c * CHc) * Dc;
  stage_bf16(P, LD64, proj, Mc, Dc, 1.0f);
  stage_bf16(Xk, LD64, kin + base, CHc, Dc, DN);
  stage_bf16_T(VT, LD128, vin + base, CHc, Dc);
  if (tid < CHc) {
    float s = 0.0f;
    for (int d = 0; d < Dc; ++d) { float x = kin[base + (size_t)tid * Dc + d]; s += x * x; }
    diagk[tid] = 0.0625f * s;   // 0.5 * ||k||^2 * D^-0.5
  }
  const float gmax = fdec(*gkey);
  __syncthreads();
  // kp = ratio*(exp(dd - diag - gmax) + eps), written transposed -> KPT[m][j]
  {
    v16bf a0 = load_afrag(Xk, LD64, w * 16, 0);
    v16bf a1 = load_afrag(Xk, LD64, w * 16, 32);
    for (int tj = 0; tj < 16; ++tj) {
      v8f acc = wmma_bf16(a0, load_bfragT(P, LD64, tj * 16, 0), vzero8());
      acc = wmma_bf16(a1, load_bfragT(P, LD64, tj * 16, 32), acc);
#pragma unroll
      for (int r = 0; r < 8; ++r) {
        int j = w * 16 + r + 8 * hi;
        int m = tj * 16 + (lane & 15);
        float kp = RATIO * (expf(acc[r] - diagk[j] - gmax) + KEPS);
        KPT[(size_t)m * LD128 + j] = (__bf16)kp;
      }
    }
  }
  __syncthreads();
  // S[m][d] = sum_j KPT[m][j] * VT[d][j]; store transposed gS[d][m]
  const size_t sb = (size_t)blockIdx.x * Mc * Dc;
  for (int mt = w; mt < 16; mt += 8) {
    v16bf a0 = load_afrag(KPT, LD128, mt * 16, 0);
    v16bf a1 = load_afrag(KPT, LD128, mt * 16, 32);
    v16bf a2 = load_afrag(KPT, LD128, mt * 16, 64);
    v16bf a3 = load_afrag(KPT, LD128, mt * 16, 96);
    for (int dt = 0; dt < 4; ++dt) {
      v8f acc = wmma_bf16(a0, load_bfragT(VT, LD128, dt * 16, 0), vzero8());
      acc = wmma_bf16(a1, load_bfragT(VT, LD128, dt * 16, 32), acc);
      acc = wmma_bf16(a2, load_bfragT(VT, LD128, dt * 16, 64), acc);
      acc = wmma_bf16(a3, load_bfragT(VT, LD128, dt * 16, 96), acc);
#pragma unroll
      for (int r = 0; r < 8; ++r) {
        int m = mt * 16 + r + 8 * hi;
        int d = dt * 16 + (lane & 15);
        gS[sb + (size_t)d * Mc + m] = (__bf16)acc[r];
      }
    }
  }
  if (tid < Mc) {
    float z = 0.0f;
    for (int j = 0; j < CHc; ++j) z += (float)KPT[(size_t)tid * LD128 + j];
    gZ[(size_t)blockIdx.x * Mc + tid] = z;
  }
}

// Pass 3: exclusive prefix scan over chunks (per b,h) for S (bf16) and Z (f32)
__global__ __launch_bounds__(256) void perf_scan_kernel(__bf16* __restrict__ gS,
                                                        float* __restrict__ gZ) {
  const int bh = blockIdx.x, tid = threadIdx.x;
  float run[64];
#pragma unroll
  for (int j = 0; j < 64; ++j) run[j] = 0.0f;
  float zr = 0.0f;
  for (int c = 0; c < NCH; ++c) {
    const size_t sb = ((size_t)(bh * NCH + c)) * Mc * Dc;
    for (int j = 0; j < 64; ++j) {
      size_t e = sb + tid + 256 * j;
      float x = (float)gS[e]; gS[e] = (__bf16)run[j]; run[j] += x;
    }
    size_t zb = ((size_t)(bh * NCH + c)) * Mc + tid;
    float z = gZ[zb]; gZ[zb] = zr; zr += z;
  }
}

// main-kernel LDS layout (bytes)
static constexpr unsigned OFF_QP  = 0;        // [128][264] bf16
static constexpr unsigned OFF_KP  = 67584;    // [128][264] bf16 (row-major [j][m])
static constexpr unsigned OFF_VT  = 135168;   // [64][136]  bf16 v^T
static constexpr unsigned OFF_SPT = 152576;   // [64][264]  bf16 S_prefix^T [d][m]
static constexpr unsigned OFF_SCR = 186368;   // scratch: P | Xq | Xk ; later SC
static constexpr unsigned OFF_P   = 186368;   // [256][72]
static constexpr unsigned OFF_XQ  = 223232;   // [128][72]
static constexpr unsigned OFF_XK  = 241664;   // [128][72]
static constexpr unsigned OFF_SC  = 186368;   // [128][136] masked scores (aliases P)
static constexpr unsigned OFF_ZP  = 260096;   // [256] f32
static constexpr unsigned OFF_RK  = 261120;   // [128] u32 row-max keys
static constexpr unsigned OFF_DQ  = 261632;   // [128] f32
static constexpr unsigned OFF_DK  = 262144;   // [128] f32
static constexpr unsigned OFF_DEN = 262656;   // [128] f32
static constexpr unsigned SM_MAIN = 263168;   // ~257 KB (< 320 KB/WGP)

// Pass 4: fused per-chunk attention
__global__ __launch_bounds__(256) void perf_main_kernel(const float* __restrict__ qin,
                                                        const float* __restrict__ kin,
                                                        const float* __restrict__ vin,
                                                        const float* __restrict__ proj,
                                                        const unsigned* gkey,
                                                        const __bf16* __restrict__ gS,
                                                        const float* __restrict__ gZ,
                                                        float* __restrict__ out) {
  extern __shared__ char smem[];
  __bf16* QP  = (__bf16*)(smem + OFF_QP);
  __bf16* KP  = (__bf16*)(smem + OFF_KP);
  __bf16* VT  = (__bf16*)(smem + OFF_VT);
  __bf16* SPT = (__bf16*)(smem + OFF_SPT);
  __bf16* P   = (__bf16*)(smem + OFF_P);
  __bf16* Xq  = (__bf16*)(smem + OFF_XQ);
  __bf16* Xk  = (__bf16*)(smem + OFF_XK);
  __bf16* SC  = (__bf16*)(smem + OFF_SC);
  float*    zp   = (float*)(smem + OFF_ZP);
  unsigned* rkey = (unsigned*)(smem + OFF_RK);
  float*    dq   = (float*)(smem + OFF_DQ);
  float*    dk   = (float*)(smem + OFF_DK);
  float*    den  = (float*)(smem + OFF_DEN);

  const int bh = blockIdx.x / NCH, c = blockIdx.x % NCH;
  const int tid = threadIdx.x, lane = tid & 31, hi = lane >> 4;
  const int w = __builtin_amdgcn_readfirstlane((int)(tid >> 5));
  const size_t base = (size_t)(bh * Nc + c * CHc) * Dc;
  const size_t sb = (size_t)blockIdx.x * Mc * Dc;
  const size_t zb = (size_t)blockIdx.x * Mc;

#if HAVE_TDM
  // TDM: raw-copy 32 KB S_prefix^T chunk into LDS, inserting the +8-bf16 row
  // padding in hardware (pad every 128 dwords by 4 dwords -> row stride 264).
  if (w == 0) {
    unsigned long long ga = (unsigned long long)(size_t)(gS + sb);
    v4u g0;
    g0.x = 1u;                                            // count=1, load, user
    g0.y = OFF_SPT;                                       // lds_addr
    g0.z = (unsigned)(ga & 0xFFFFFFFFu);                  // global_addr[31:0]
    g0.w = (unsigned)((ga >> 32) & 0x01FFFFFFu) | 0x80000000u; // addr[56:32], type=2
    v8i g1;
    g1[0] = (int)((1u << 16) | (1u << 20) | (6u << 22) | (3u << 25));
    //        data_size=2B    pad_enable   interval=128dw  amount=4dw
    g1[1] = (int)(16384u << 16);   // tensor_dim0 = 16384 elements (bits 79:48)
    g1[2] = (int)(1u << 16);       // tensor_dim0 hi=0; tensor_dim1 = 1
    g1[3] = (int)(16384u << 16);   // tensor_dim1 hi=0; tile_dim0 = 16384
    g1[4] = 1;                     // tile_dim1 = 1; tile_dim2 = 0
    g1[5] = 16384;                 // tensor_dim0_stride
    g1[6] = 0;
    g1[7] = 0;
    v4i zg4 = {0, 0, 0, 0};
    v8i zg8 = {0, 0, 0, 0, 0, 0, 0, 0};
    __builtin_amdgcn_tensor_load_to_lds(g0, g1, zg4, zg4, zg8, 0);
  }
#else
  for (int i = tid; i < Mc * Dc; i += 256) {
    int d = i >> 8, m = i & 255;
    SPT[(size_t)d * LD256 + m] = gS[sb + i];
  }
#endif

  stage_bf16(P, LD64, proj, Mc, Dc, 1.0f);
  stage_bf16(Xq, LD64, qin + base, CHc, Dc, DN);
  stage_bf16(Xk, LD64, kin + base, CHc, Dc, DN);
  stage_bf16_T(VT, LD128, vin + base, CHc, Dc);
  for (int i = tid; i < Mc; i += 256) zp[i] = gZ[zb + i];
  if (tid < CHc) {
    float sq = 0.0f, sk = 0.0f;
    for (int d = 0; d < Dc; ++d) {
      float x = qin[base + (size_t)tid * Dc + d]; sq += x * x;
      float y = kin[base + (size_t)tid * Dc + d]; sk += y * y;
    }
    dq[tid] = 0.0625f * sq;
    dk[tid] = 0.0625f * sk;
    rkey[tid] = 0u;
    den[tid] = 0.0f;
  }
  const float gmax = fdec(*gkey);
#if HAVE_TDM
  if (w == 0) __builtin_amdgcn_s_wait_tensorcnt(0);
#endif
  __syncthreads();

  // wave w owns rows [w*16, w*16+16) throughout
  v16bf xq0 = load_afrag(Xq, LD64, w * 16, 0);
  v16bf xq1 = load_afrag(Xq, LD64, w * 16, 32);
  v16bf xk0 = load_afrag(Xk, LD64, w * 16, 0);
  v16bf xk1 = load_afrag(Xk, LD64, w * 16, 32);

  // q projection pass A: per-row max of data_dash
  for (int tj = 0; tj < 16; ++tj) {
    v8f acc = wmma_bf16(xq0, load_bfragT(P, LD64, tj * 16, 0), vzero8());
    acc = wmma_bf16(xq1, load_bfragT(P, LD64, tj * 16, 32), acc);
#pragma unroll
    for (int r = 0; r < 8; ++r)
      atomicMax(&rkey[w * 16 + r + 8 * hi], fkey(acc[r]));
  }
  __syncthreads();

  // q pass B (row stabilizer) and k pass (global stabilizer)
  for (int tj = 0; tj < 16; ++tj) {
    v16bf b0 = load_bfragT(P, LD64, tj * 16, 0);
    v16bf b1 = load_bfragT(P, LD64, tj * 16, 32);
    v8f aq = wmma_bf16(xq0, b0, vzero8()); aq = wmma_bf16(xq1, b1, aq);
    v8f ak = wmma_bf16(xk0, b0, vzero8()); ak = wmma_bf16(xk1, b1, ak);
#pragma unroll
    for (int r = 0; r < 8; ++r) {
      int i = w * 16 + r + 8 * hi;
      int m = tj * 16 + (lane & 15);
      float qv = RATIO * (expf(aq[r] - dq[i] - fdec(rkey[i])) + KEPS);
      float kv = RATIO * (expf(ak[r] - dk[i] - gmax) + KEPS);
      QP[(size_t)i * LD256 + m] = (__bf16)qv;
      KP[(size_t)i * LD256 + m] = (__bf16)kv;
    }
  }
  __syncthreads();

  // hoisted A fragments of qp rows (reused for scores and num_inter)
  v16bf qa[8];
#pragma unroll
  for (int s = 0; s < 8; ++s) qa[s] = load_afrag(QP, LD256, w * 16, s * 32);

  // scores = qp kp^T (K=256), causal mask, rowsum -> den_intra
  for (int tj = 0; tj < 8; ++tj) {
    v8f acc = vzero8();
#pragma unroll
    for (int s = 0; s < 8; ++s)
      acc = wmma_bf16(qa[s], load_bfragT(KP, LD256, tj * 16, s * 32), acc);
#pragma unroll
    for (int r = 0; r < 8; ++r) {
      int i = w * 16 + r + 8 * hi;
      int j = tj * 16 + (lane & 15);
      float s = (j <= i) ? acc[r] : 0.0f;
      SC[(size_t)i * LD128 + j] = (__bf16)s;
      if (j <= i) atomicAdd(&den[i], s);
    }
  }
  __syncthreads();

  // den += sum_m qp[i][m]*(Z_prefix[m] + eps)  (den_inter + eps*rowsum(qp))
  if (tid < CHc) {
    float s = 0.0f;
    for (int m = 0; m < Mc; ++m)
      s += (float)QP[(size_t)tid * LD256 + m] * (zp[m] + EPSC);
    den[tid] += s;
  }
  __syncthreads();

  // num = scores*V + qp*S_prefix ; out = num / den
  v16bf sa[4];
#pragma unroll
  for (int s = 0; s < 4; ++s) sa[s] = load_afrag(SC, LD128, w * 16, s * 32);
  for (int dt = 0; dt < 4; ++dt) {
    v8f acc = vzero8();
#pragma unroll
    for (int s = 0; s < 4; ++s)
      acc = wmma_bf16(sa[s], load_bfragT(VT, LD128, dt * 16, s * 32), acc);
#pragma unroll
    for (int s = 0; s < 8; ++s)
      acc = wmma_bf16(qa[s], load_bfragT(SPT, LD256, dt * 16, s * 32), acc);
#pragma unroll
    for (int r = 0; r < 8; ++r) {
      int i = w * 16 + r + 8 * hi;
      int d = dt * 16 + (lane & 15);
      out[base + (size_t)i * Dc + d] = acc[r] / den[i];
    }
  }
}

// ---------------------------------------------------------------------------
extern "C" void kernel_launch(void* const* d_in, const int* in_sizes, int n_in,
                              void* d_out, int out_size, void* d_ws, size_t ws_size,
                              hipStream_t stream) {
  (void)in_sizes; (void)n_in; (void)out_size; (void)ws_size;
  const float* q    = (const float*)d_in[0];
  const float* k    = (const float*)d_in[1];
  const float* v    = (const float*)d_in[2];
  const float* proj = (const float*)d_in[3];
  float* out = (float*)d_out;

  char* ws = (char*)d_ws;
  __bf16*   gS   = (__bf16*)ws;
  float*    gZ   = (float*)(ws + S_BYTES);
  unsigned* gkey = (unsigned*)(ws + S_BYTES + Z_BYTES);

  const size_t SM_KMAX = 36864 + 18432;                    // ~54 KB
  const size_t SM_SZ   = 69632 + 17408 + 36864 + 18432 + 512; // ~140 KB

  perf_init_kernel<<<1, 1, 0, stream>>>(gkey);
  perf_kmax_kernel<<<NBLK, 256, SM_KMAX, stream>>>(k, proj, gkey);
  perf_sz_kernel<<<NBLK, 256, SM_SZ, stream>>>(k, v, proj, gkey, gS, gZ);
  perf_scan_kernel<<<BH, 256, 0, stream>>>(gS, gZ);
  perf_main_kernel<<<NBLK, 256, SM_MAIN, stream>>>(q, k, v, proj, gkey, gS, gZ, out);
}
